// attention_18210661335624
// MI455X (gfx1250) — matrix operand order repaired and checked by reading, not compiled, not run on hardware
//
#include <hip/hip_runtime.h>

// Problem constants (match reference)
#define Bb   8
#define Tt   2048
#define Dd   1024
#define DOUT 1024

typedef __bf16 bf16_t;
typedef bf16_t v16bf __attribute__((ext_vector_type(16)));
typedef bf16_t v8bf  __attribute__((ext_vector_type(8)));
typedef float  v8f   __attribute__((ext_vector_type(8)));

// ---------------------------------------------------------------------------
// CDNA5 async global->LDS copy (ISA 10.x / 15.18.3, GVS mode):
//   global_load_async_to_lds_b128 vdst(lds addr), vaddr(32b byte off), saddr
// Tracked by ASYNCcnt; drained with s_wait_asynccnt before the barrier.
// Inline asm (portable across both toolchains' builtin arities).
// ---------------------------------------------------------------------------
__device__ __forceinline__ unsigned lds_addr32(const void* p) {
    return (unsigned)(uintptr_t)p;   // flat LDS aperture: addr[31:0] == LDS byte addr
}
__device__ __forceinline__ void async_copy_b128(unsigned ldsA, const void* gbase,
                                                unsigned gByteOff) {
    asm volatile("global_load_async_to_lds_b128 %0, %1, %2"
                 :: "v"(ldsA), "v"(gByteOff), "s"(gbase) : "memory");
}
__device__ __forceinline__ void wait_async0() {
    asm volatile("s_wait_asynccnt 0x0" ::: "memory");
}

// ---------------------------------------------------------------------------
// WMMA fragment loaders, CDNA5 ISA 7.12.2 VGPR layouts (wave32):
//  A 16x32 bf16 : lane L -> M=L&15, h=L>>4; elems = K[8h..8h+8) ++ K[16+8h..+8)
//  B 32x16 bf16 : lane L -> N=L&15, h=L>>4; elems = K[8h..8h+8) ++ K[16+8h..+8)
// LDS tiles: A stored [M][32], B stored [N][32] -> two ds_load_b128 per frag.
// ---------------------------------------------------------------------------
__device__ __forceinline__ v16bf ld_a_frag(const bf16_t* At, int rowBase, int lane) {
    int m = rowBase + (lane & 15);
    int h = lane >> 4;
    v8bf lo = *(const v8bf*)(At + m * 32 + 8 * h);
    v8bf hi = *(const v8bf*)(At + m * 32 + 16 + 8 * h);
    return __builtin_shufflevector(lo, hi, 0,1,2,3,4,5,6,7,8,9,10,11,12,13,14,15);
}
__device__ __forceinline__ v16bf ld_b_frag(const bf16_t* Bt, int colBase, int lane) {
    int n = colBase + (lane & 15);
    int h = lane >> 4;
    v8bf lo = *(const v8bf*)(Bt + n * 32 + 8 * h);
    v8bf hi = *(const v8bf*)(Bt + n * 32 + 16 + 8 * h);
    return __builtin_shufflevector(lo, hi, 0,1,2,3,4,5,6,7,8,9,10,11,12,13,14,15);
}
__device__ __forceinline__ v8f wmma_bf16(v16bf a, v16bf b, v8f c) {
    return __builtin_amdgcn_wmma_f32_16x16x32_bf16(false, a, false, b, (short)0, c,
                                                   false, false);
}
__device__ __forceinline__ void store4_bf16(bf16_t* dst, float a, float b,
                                            float c, float d) {
    union { uint2 u; bf16_t h[4]; } t;
    t.h[0] = (bf16_t)a; t.h[1] = (bf16_t)b; t.h[2] = (bf16_t)c; t.h[3] = (bf16_t)d;
    *(uint2*)dst = t.u;
}

// ---------------------------------------------------------------------------
// k_zero: clear colsum accumulator (must run every launch)
// ---------------------------------------------------------------------------
__global__ void k_zero(float* __restrict__ p, int n) {
    int i = blockIdx.x * 256 + threadIdx.x;
    if (i < n) p[i] = 0.0f;
}

// ---------------------------------------------------------------------------
// k_proj: Xp = bf16(x @ W + bias). 128x128 tile/block, double-buffered LDS,
// two-phase (load-early / convert+store-late) pipeline around the WMMAs.
// ---------------------------------------------------------------------------
__global__ __launch_bounds__(256) void k_proj(const float* __restrict__ x,
                                              const float* __restrict__ W,
                                              const float* __restrict__ bias,
                                              bf16_t* __restrict__ Xp) {
    __shared__ __align__(16) bf16_t At[2][128 * 32];
    __shared__ __align__(16) bf16_t Bt[2][128 * 32];
    const int tid = threadIdx.x, lane = tid & 31, wid = tid >> 5;
    const int waveM = wid & 3, waveN = wid >> 2;
    const int mBase = blockIdx.y * 128, nBase = blockIdx.x * 128;

    float4 aReg[4], bReg[4];
    auto loadRegs = [&](int kc) {
        for (int i = 0; i < 4; ++i) {
            int idx = tid + i * 256;
            int row = idx >> 3, c4 = (idx & 7) * 4;           // A: 128x32
            aReg[i] = *(const float4*)(x + (size_t)(mBase + row) * Dd + kc + c4);
            int k = idx >> 5, n4 = (idx & 31) * 4;            // B: 32x128
            bReg[i] = *(const float4*)(W + (size_t)(kc + k) * DOUT + nBase + n4);
        }
    };
    auto storeTiles = [&](int buf) {
        for (int i = 0; i < 4; ++i) {
            int idx = tid + i * 256;
            int row = idx >> 3, c4 = (idx & 7) * 4;
            store4_bf16(&At[buf][row * 32 + c4], aReg[i].x, aReg[i].y, aReg[i].z, aReg[i].w);
            int k = idx >> 5, n4 = (idx & 31) * 4;            // transpose into [n][k]
            Bt[buf][(n4 + 0) * 32 + k] = (bf16_t)bReg[i].x;
            Bt[buf][(n4 + 1) * 32 + k] = (bf16_t)bReg[i].y;
            Bt[buf][(n4 + 2) * 32 + k] = (bf16_t)bReg[i].z;
            Bt[buf][(n4 + 3) * 32 + k] = (bf16_t)bReg[i].w;
        }
    };

    v8f acc[2][4] = {};
    loadRegs(0); storeTiles(0); __syncthreads();
    int cur = 0;
    for (int kc = 0; kc < Dd; kc += 32) {
        const bool nxt = (kc + 32 < Dd);
        if (nxt) loadRegs(kc + 32);                  // global loads in flight
        v16bf af[2], bfr[4];
        for (int i = 0; i < 2; ++i) af[i]  = ld_a_frag(At[cur], waveM * 32 + i * 16, lane);
        for (int j = 0; j < 4; ++j) bfr[j] = ld_b_frag(Bt[cur], waveN * 64 + j * 16, lane);
        for (int i = 0; i < 2; ++i)
            for (int j = 0; j < 4; ++j)
                acc[i][j] = wmma_bf16(af[i], bfr[j], acc[i][j]);
        if (nxt) storeTiles(cur ^ 1);                // cvt+ds_store after WMMAs
        __syncthreads();
        cur ^= 1;
    }
    const int nl = lane & 15, h = lane >> 4;
    for (int j = 0; j < 4; ++j) {
        int n = nBase + waveN * 64 + j * 16 + nl;
        float bj = bias[n];
        for (int i = 0; i < 2; ++i)
            for (int r = 0; r < 8; ++r) {
                int m = mBase + waveM * 32 + i * 16 + r + 8 * h;
                Xp[(size_t)m * DOUT + n] = (bf16_t)(acc[i][j][r] + bj);
            }
    }
}

// ---------------------------------------------------------------------------
// k_scores: S = Xp_t @ Xp_s^T / 32 on lower-triangular tiles; E = exp (bf16),
// colsum[s] += column sums. Both LDS tiles are pure bf16 copies -> async DMA.
// ---------------------------------------------------------------------------
__global__ __launch_bounds__(256) void k_scores(const bf16_t* __restrict__ Xp,
                                                bf16_t* __restrict__ E,
                                                float* __restrict__ colsum) {
    const int sTile = blockIdx.x, tTile = blockIdx.y, b = blockIdx.z;
    if (tTile < sTile) return;   // block-uniform; masked tiles never read
    __shared__ __align__(16) bf16_t At[2][128 * 32];
    __shared__ __align__(16) bf16_t Bt[2][128 * 32];
    const int tid = threadIdx.x, lane = tid & 31, wid = tid >> 5;
    const int waveM = wid & 3, waveN = wid >> 2;
    const int tBase = tTile * 128, sBase = sTile * 128;
    const bf16_t* Xb = Xp + (size_t)b * Tt * DOUT;

    auto issue = [&](int buf, int kc) {              // 8KB per tile, 16B/lane chunks
        for (int i = 0; i < 2; ++i) {
            int idx = tid + i * 256;
            int row = idx >> 2, c8 = (idx & 3) * 8;
            async_copy_b128(lds_addr32(&At[buf][row * 32 + c8]), Xb,
                            (unsigned)(((tBase + row) * DOUT + kc + c8) * 2));
            async_copy_b128(lds_addr32(&Bt[buf][row * 32 + c8]), Xb,
                            (unsigned)(((sBase + row) * DOUT + kc + c8) * 2));
        }
    };

    v8f acc[2][4] = {};
    issue(0, 0); wait_async0(); __syncthreads();
    int cur = 0;
    for (int kc = 0; kc < DOUT; kc += 32) {
        if (kc + 32 < DOUT) issue(cur ^ 1, kc + 32); // async DMA overlaps WMMAs
        v16bf af[2], bfr[4];
        for (int i = 0; i < 2; ++i) af[i]  = ld_a_frag(At[cur], waveM * 32 + i * 16, lane);
        for (int j = 0; j < 4; ++j) bfr[j] = ld_b_frag(Bt[cur], waveN * 64 + j * 16, lane);
        for (int i = 0; i < 2; ++i)
            for (int j = 0; j < 4; ++j)
                acc[i][j] = wmma_bf16(af[i], bfr[j], acc[i][j]);
        wait_async0();
        __syncthreads();
        cur ^= 1;
    }
    const int nl = lane & 15, h = lane >> 4;
    bf16_t* Eb = E + (size_t)b * Tt * Tt;
    for (int j = 0; j < 4; ++j) {
        int s = sBase + waveN * 64 + j * 16 + nl;    // same s for lanes L, L+16
        float part = 0.0f;
        for (int i = 0; i < 2; ++i)
            for (int r = 0; r < 8; ++r) {
                int t = tBase + waveM * 32 + i * 16 + r + 8 * h;
                float v = (t >= s) ? __expf(acc[i][j][r] * 0.03125f) : 0.0f;
                Eb[(size_t)t * Tt + s] = (bf16_t)v;
                part += v;
            }
        part += __shfl_xor(part, 16, 32);
        if (lane < 16) atomicAdd(&colsum[b * Tt + s], part);
    }
}

// ---------------------------------------------------------------------------
// k_scale: Xn[b,s,d] = bf16( x[b,s,d] * rcp(colsum[b,s]) )  (done once; k_out
// would otherwise redo this scaling up to 16x per element)
// ---------------------------------------------------------------------------
__global__ __launch_bounds__(256) void k_scale(const float* __restrict__ x,
                                               const float* __restrict__ colsum,
                                               bf16_t* __restrict__ Xn) {
    size_t i = ((size_t)blockIdx.x * 256 + threadIdx.x) * 4;
    int row = (int)(i >> 10);                        // /Dd : flat [B*T] row
    float r = __builtin_amdgcn_rcpf(colsum[row]);
    float4 v = *(const float4*)(x + i);
    store4_bf16(Xn + i, v.x * r, v.y * r, v.z * r, v.w * r);
}

// ---------------------------------------------------------------------------
// k_out: out[t,d] = sum_{s<=t} E[t,s] * Xn[s,d].  A tile = async bf16 copy of
// E rows; B tile = bf16 load + LDS transpose (two-phase). Flattened causal K.
// ---------------------------------------------------------------------------
__global__ __launch_bounds__(256) void k_out(const bf16_t* __restrict__ E,
                                             const bf16_t* __restrict__ Xn,
                                             float* __restrict__ out) {
    const int dTile = blockIdx.x, tTile = blockIdx.y, b = blockIdx.z;
    __shared__ __align__(16) bf16_t At[2][128 * 32];
    __shared__ __align__(16) bf16_t Bt[2][128 * 32];
    const int tid = threadIdx.x, lane = tid & 31, wid = tid >> 5;
    const int waveM = wid & 3, waveN = wid >> 2;
    const int tBase = tTile * 128, dBase = dTile * 128;
    const bf16_t* Eb  = E + (size_t)b * Tt * Tt;
    const bf16_t* Xnb = Xn + (size_t)b * Tt * Dd;

    auto issueA = [&](int buf, int ki) {
        int sOff = ki * 32;
        for (int i = 0; i < 2; ++i) {
            int idx = tid + i * 256;
            int row = idx >> 2, c8 = (idx & 3) * 8;
            async_copy_b128(lds_addr32(&At[buf][row * 32 + c8]), Eb,
                            (unsigned)(((size_t)(tBase + row) * Tt + sOff + c8) * 2));
        }
    };
    uint2 bReg[4];
    auto loadB = [&](int ki) {                       // 32(s) x 128(d), d-contiguous
        int sOff = ki * 32;
        for (int i = 0; i < 4; ++i) {
            int idx = tid + i * 256;
            int k = idx >> 5, d4 = (idx & 31) * 4;
            bReg[i] = *(const uint2*)(Xnb + (size_t)(sOff + k) * Dd + dBase + d4);
        }
    };
    auto storeB = [&](int buf) {                     // transpose into [d][s]
        for (int i = 0; i < 4; ++i) {
            int idx = tid + i * 256;
            int k = idx >> 5, d4 = (idx & 31) * 4;
            union { uint2 u; bf16_t h[4]; } t; t.u = bReg[i];
            Bt[buf][(d4 + 0) * 32 + k] = t.h[0];
            Bt[buf][(d4 + 1) * 32 + k] = t.h[1];
            Bt[buf][(d4 + 2) * 32 + k] = t.h[2];
            Bt[buf][(d4 + 3) * 32 + k] = t.h[3];
        }
    };

    v8f acc[2][4] = {};
    const int nk = (tTile + 1) * 4;                  // causal: s <= t tiles only
    issueA(0, 0); loadB(0); storeB(0); wait_async0(); __syncthreads();
    int cur = 0;
    for (int ki = 0; ki < nk; ++ki) {
        const bool nxt = (ki + 1 < nk);
        if (nxt) { issueA(cur ^ 1, ki + 1); loadB(ki + 1); }
        v16bf af[2], bfr[4];
        for (int i = 0; i < 2; ++i) af[i]  = ld_a_frag(At[cur], waveM * 32 + i * 16, lane);
        for (int j = 0; j < 4; ++j) bfr[j] = ld_b_frag(Bt[cur], waveN * 64 + j * 16, lane);
        for (int i = 0; i < 2; ++i)
            for (int j = 0; j < 4; ++j)
                acc[i][j] = wmma_bf16(af[i], bfr[j], acc[i][j]);
        if (nxt) storeB(cur ^ 1);
        wait_async0();
        __syncthreads();
        cur ^= 1;
    }
    const int nl = lane & 15, h = lane >> 4;
    for (int j = 0; j < 4; ++j) {
        int n = dBase + waveN * 64 + j * 16 + nl;
        for (int i = 0; i < 2; ++i)
            for (int r = 0; r < 8; ++r) {
                int t = tBase + waveM * 32 + i * 16 + r + 8 * h;
                out[(size_t)b * Tt * Dd + (size_t)t * Dd + n] = acc[i][j][r];
            }
    }
}

// ---------------------------------------------------------------------------
// ws layout: [ Xp bf16 32MB | E bf16 64MB | Xn bf16 32MB | colsum f32 64KB ]
// ---------------------------------------------------------------------------
extern "C" void kernel_launch(void* const* d_in, const int* in_sizes, int n_in,
                              void* d_out, int out_size, void* d_ws, size_t ws_size,
                              hipStream_t stream) {
    (void)in_sizes; (void)n_in; (void)out_size; (void)ws_size;
    const float* x    = (const float*)d_in[0];
    const float* W    = (const float*)d_in[1];
    const float* bias = (const float*)d_in[2];
    float* out = (float*)d_out;

    char* ws = (char*)d_ws;
    const size_t xpBytes = (size_t)Bb * Tt * DOUT * sizeof(bf16_t);
    const size_t eBytes  = (size_t)Bb * Tt * Tt * sizeof(bf16_t);
    const size_t xnBytes = (size_t)Bb * Tt * Dd * sizeof(bf16_t);
    bf16_t* Xp     = (bf16_t*)ws;
    bf16_t* E      = (bf16_t*)(ws + xpBytes);
    bf16_t* Xn     = (bf16_t*)(ws + xpBytes + eBytes);
    float*  colsum = (float*)(ws + xpBytes + eBytes + xnBytes);

    k_zero<<<(Bb * Tt + 255) / 256, 256, 0, stream>>>(colsum, Bb * Tt);
    k_proj<<<dim3(DOUT / 128, (Bb * Tt) / 128), 256, 0, stream>>>(x, W, bias, Xp);
    k_scores<<<dim3(Tt / 128, Tt / 128, Bb), 256, 0, stream>>>(Xp, E, colsum);
    k_scale<<<(Bb * Tt * Dd) / 1024, 256, 0, stream>>>(x, colsum, Xn);
    k_out<<<dim3(Dd / 128, Tt / 128, Bb), 256, 0, stream>>>(E, Xn, out);
}
